// GNNQNetwork_32452772888805
// MI455X (gfx1250) — matrix operand order repaired and checked by reading, not compiled, not run on hardware
//
#include <hip/hip_runtime.h>
#include <hip/hip_bf16.h>

// ---------------------------------------------------------------------------
// Types for WMMA
// ---------------------------------------------------------------------------
typedef __bf16 bf16x16 __attribute__((ext_vector_type(16)));
typedef __bf16 bf16x8  __attribute__((ext_vector_type(8)));
typedef float  f32x8   __attribute__((ext_vector_type(8)));

union ABfrag { bf16x16 v; bf16x8 h[2]; };

__device__ __forceinline__ unsigned short f2bf(float f) {
    unsigned u = __float_as_uint(f);
    u += 0x7FFFu + ((u >> 16) & 1u);           // round-to-nearest-even
    return (unsigned short)(u >> 16);
}

// ---------------------------------------------------------------------------
// CDNA5 async global->LDS copy (ASYNCcnt-tracked). LDS byte address is the
// low 32 bits of the generic pointer (flat LDS aperture maps addr[31:0]).
// ---------------------------------------------------------------------------
#define USE_ASYNC_LDS 1

__device__ __forceinline__ void async_copy_b128(const unsigned short* g, unsigned short* l) {
    unsigned laddr = (unsigned)(uintptr_t)l;
    asm volatile("global_load_async_to_lds_b128 %0, %1, off"
                 :: "v"(laddr), "v"(g)
                 : "memory");
}
__device__ __forceinline__ void wait_asynccnt0() {
    asm volatile("s_wait_asynccnt 0" ::: "memory");
}

// ---------------------------------------------------------------------------
// Utility kernels
// ---------------------------------------------------------------------------
__global__ void cast_bf16_k(const float* __restrict__ src, unsigned short* __restrict__ dst, long n) {
    long i = (long)blockIdx.x * blockDim.x + threadIdx.x;
    long stride = (long)gridDim.x * blockDim.x;
    for (; i < n; i += stride) dst[i] = f2bf(src[i]);
}

// Transpose-cast weights: dst[n*K + k] = bf16(src[k*Nout + n])  (dst is [Nout][K])
__global__ void cast_bf16_t_k(const float* __restrict__ src, unsigned short* __restrict__ dst,
                              int K, int Nout) {
    int i = blockIdx.x * blockDim.x + threadIdx.x;
    if (i >= K * Nout) return;
    int n = i / K, k = i % K;
    dst[i] = f2bf(src[(size_t)k * Nout + n]);
}

__global__ void zero_f32_k(float* __restrict__ p, long n) {
    long i = (long)blockIdx.x * blockDim.x + threadIdx.x;
    long stride = (long)gridDim.x * blockDim.x;
    for (; i < n; i += stride) p[i] = 0.0f;
}

__global__ void init_softmax_k(float* __restrict__ m, float* __restrict__ den, int n) {
    int i = blockIdx.x * blockDim.x + threadIdx.x;
    if (i < n) { m[i] = -__builtin_inff(); den[i] = 0.0f; }
}

// ---------------------------------------------------------------------------
// bf16 WMMA GEMM: C[M,Nout] = A[M,K] * BT[Nout,K]^T, A/BT bf16, C f32.
// Block = 128 threads (4 waves), workgroup tile 64x64, wave tile 32x32
// via 2x2 v_wmma_f32_16x16x32_bf16, K stepped by 32 through LDS.
// Tiles staged with global_load_async_to_lds_b128 (2 fixed 16B chunks per
// thread per tile); ragged last M-block uses guarded sync loads.
// ---------------------------------------------------------------------------
#define GTM 64
#define GTN 64
#define GTK 32
#define LSTR 40   // LDS row stride in bf16 elems (80B, 16B-aligned, bank-spread)

__launch_bounds__(128)
__global__ void gemm_bf16_wmma(const unsigned short* __restrict__ A,
                               const unsigned short* __restrict__ BT,
                               float* __restrict__ C,
                               int M, int K, int Nout) {
    __shared__ __align__(16) unsigned short ldsA[GTM * LSTR];
    __shared__ __align__(16) unsigned short ldsB[GTN * LSTR];

    const int t    = threadIdx.x;
    const int lane = t & 31;
    const int wv   = t >> 5;        // wave 0..3
    const int wm   = wv >> 1;       // wave row 0..1
    const int wn   = wv & 1;        // wave col 0..1
    const int hlf  = (lane >> 4) & 1;
    const int l16  = lane & 15;
    const int bm   = blockIdx.x * GTM;
    const int bn   = blockIdx.y * GTN;

    // Each thread owns chunk rows (r0, r0+32) at 16B chunk `part` of each tile.
    const int r0   = t >> 2;        // 0..31
    const int part = t & 3;         // 0..3  (8 bf16 = 16B)
    const bool fullM = (bm + GTM) <= M;

    unsigned short* la0 = ldsA + r0 * LSTR + part * 8;
    unsigned short* la1 = ldsA + (r0 + 32) * LSTR + part * 8;
    unsigned short* lb0 = ldsB + r0 * LSTR + part * 8;
    unsigned short* lb1 = ldsB + (r0 + 32) * LSTR + part * 8;

    const unsigned short* gA0 = A + (size_t)(bm + r0) * K + part * 8;
    const unsigned short* gA1 = A + (size_t)(bm + r0 + 32) * K + part * 8;
    const unsigned short* gB0 = BT + (size_t)(bn + r0) * K + part * 8;
    const unsigned short* gB1 = BT + (size_t)(bn + r0 + 32) * K + part * 8;

    f32x8 acc[2][2];
    for (int i = 0; i < 2; ++i)
        for (int j = 0; j < 2; ++j)
            acc[i][j] = (f32x8){0.f, 0.f, 0.f, 0.f, 0.f, 0.f, 0.f, 0.f};

    for (int k0 = 0; k0 < K; k0 += GTK) {
#if USE_ASYNC_LDS
        if (fullM) {
            async_copy_b128(gA0 + k0, la0);
            async_copy_b128(gA1 + k0, la1);
            async_copy_b128(gB0 + k0, lb0);
            async_copy_b128(gB1 + k0, lb1);
            wait_asynccnt0();
        } else
#endif
        {
            uint4 a0 = {0u,0u,0u,0u}, a1 = {0u,0u,0u,0u};
            if (bm + r0 < M)      a0 = *(const uint4*)(gA0 + k0);
            if (bm + r0 + 32 < M) a1 = *(const uint4*)(gA1 + k0);
            uint4 b0 = *(const uint4*)(gB0 + k0);
            uint4 b1 = *(const uint4*)(gB1 + k0);
            *(uint4*)la0 = a0;
            *(uint4*)la1 = a1;
            *(uint4*)lb0 = b0;
            *(uint4*)lb1 = b1;
        }
        __syncthreads();

        // --- build fragments per ISA 16-bit layouts ---
        ABfrag afr[2], bfr[2];
        for (int i = 0; i < 2; ++i) {
            int r = wm * 32 + i * 16 + l16;
            afr[i].h[0] = *(const bf16x8*)(ldsA + r * LSTR + 8 * hlf);        // k = 8*hlf + 0..7
            afr[i].h[1] = *(const bf16x8*)(ldsA + r * LSTR + 16 + 8 * hlf);   // k = 16 + 8*hlf + 0..7
            int nl = wn * 32 + i * 16 + l16;
            bfr[i].h[0] = *(const bf16x8*)(ldsB + nl * LSTR + 8 * hlf);       // k = 8*hlf + 0..7
            bfr[i].h[1] = *(const bf16x8*)(ldsB + nl * LSTR + 16 + 8 * hlf);  // k = 16 + 8*hlf + 0..7
        }
        for (int i = 0; i < 2; ++i)
            for (int j = 0; j < 2; ++j)
                acc[i][j] = __builtin_amdgcn_wmma_f32_16x16x32_bf16(
                    false, afr[i].v, false, bfr[j].v, (short)0, acc[i][j], false, false);
        __syncthreads();
    }

    // --- store C: VGPR r -> row (r + 8*hlf), col = lane%16 ---
    for (int i = 0; i < 2; ++i)
        for (int j = 0; j < 2; ++j) {
            int col = bn + wn * 32 + j * 16 + l16;
            for (int r = 0; r < 8; ++r) {
                int row = bm + wm * 32 + i * 16 + hlf * 8 + r;
                if (row < M) C[(size_t)row * Nout + col] = acc[i][j][r];
            }
        }
}

// ---------------------------------------------------------------------------
// Attention scores: asrc[n,h] = <H[n,h,:], a_s[h,:]>, adst likewise. Wave/node.
// ---------------------------------------------------------------------------
__global__ void attn_scores_k(const float* __restrict__ H,
                              const float* __restrict__ as,
                              const float* __restrict__ ad,
                              float* __restrict__ asrc, float* __restrict__ adst,
                              int n, int heads) {
    int w    = (blockIdx.x * blockDim.x + threadIdx.x) >> 5;
    int lane = threadIdx.x & 31;
    if (w >= n) return;
    int C = heads * 128;
    for (int h = 0; h < heads; ++h) {
        float4 v = *(const float4*)(H + (size_t)w * C + h * 128 + lane * 4);
        float4 a = *(const float4*)(as + h * 128 + lane * 4);
        float4 b = *(const float4*)(ad + h * 128 + lane * 4);
        float ss = v.x * a.x + v.y * a.y + v.z * a.z + v.w * a.w;
        float dd = v.x * b.x + v.y * b.y + v.z * b.z + v.w * b.w;
        for (int o = 16; o; o >>= 1) {
            ss += __shfl_down(ss, o, 32);
            dd += __shfl_down(dd, o, 32);
        }
        if (lane == 0) {
            asrc[(size_t)w * heads + h] = ss;
            adst[(size_t)w * heads + h] = dd;
        }
    }
}

// ---------------------------------------------------------------------------
// Edge softmax passes. Self loops appended: e in [E, E+N) -> src=dst=e-E.
// ---------------------------------------------------------------------------
__device__ __forceinline__ void get_edge(const int* __restrict__ ei, int E, int e,
                                         int& s, int& d) {
    if (e < E) { s = ei[e]; d = ei[E + e]; }
    else       { s = e - E; d = e - E; }
}

__device__ __forceinline__ void atomicMaxF(float* addr, float val) {
    if (val >= 0.0f) atomicMax((int*)addr, __float_as_int(val));
    else             atomicMin((unsigned int*)addr, __float_as_uint(val));
}

__device__ __forceinline__ float lrelu02(float x) { return x > 0.0f ? x : 0.2f * x; }

__global__ void edge_max_k(const int* __restrict__ ei, int E, int Etot,
                           const float* __restrict__ asrc, const float* __restrict__ adst,
                           float* __restrict__ m, int heads) {
    int e = blockIdx.x * blockDim.x + threadIdx.x;
    if (e >= Etot) return;
    int s, d; get_edge(ei, E, e, s, d);
    for (int h = 0; h < heads; ++h) {
        float el = lrelu02(asrc[(size_t)s * heads + h] + adst[(size_t)d * heads + h]);
        atomicMaxF(&m[(size_t)d * heads + h], el);
    }
}

__global__ void edge_sum_k(const int* __restrict__ ei, int E, int Etot,
                           const float* __restrict__ asrc, const float* __restrict__ adst,
                           const float* __restrict__ m, float* __restrict__ den, int heads) {
    int e = blockIdx.x * blockDim.x + threadIdx.x;
    if (e >= Etot) return;
    int s, d; get_edge(ei, E, e, s, d);
    for (int h = 0; h < heads; ++h) {
        float el = lrelu02(asrc[(size_t)s * heads + h] + adst[(size_t)d * heads + h]);
        atomicAdd(&den[(size_t)d * heads + h], __expf(el - m[(size_t)d * heads + h]));
    }
}

// Wave per edge: lanes cover 128 floats per iter (float4 each), atomic scatter-add.
__global__ void edge_agg_k(const int* __restrict__ ei, int E, int Etot,
                           const float* __restrict__ H,
                           const float* __restrict__ asrc, const float* __restrict__ adst,
                           const float* __restrict__ m, const float* __restrict__ den,
                           float* __restrict__ agg, int heads, int C) {
    int wavesPerBlk = blockDim.x >> 5;
    int w    = blockIdx.x * wavesPerBlk + (threadIdx.x >> 5);
    int lane = threadIdx.x & 31;
    int totW = gridDim.x * wavesPerBlk;
    int iters = C >> 7;
    for (int e = w; e < Etot; e += totW) {
        int s, d; get_edge(ei, E, e, s, d);
        if (e + totW < Etot) {  // prefetch next source row (global_prefetch_b8)
            int ns, nd; get_edge(ei, E, e + totW, ns, nd);
            __builtin_prefetch((const void*)(H + (size_t)ns * C), 0, 1);
            (void)nd;
        }
        for (int it = 0; it < iters; ++it) {
            int f = it * 128 + lane * 4;
            int h = f >> 7;  // out=128 per head
            float el = lrelu02(asrc[(size_t)s * heads + h] + adst[(size_t)d * heads + h]);
            float ee = __expf(el - m[(size_t)d * heads + h]);
            float alpha = ee / (den[(size_t)d * heads + h] + 1e-16f);
            float4 v = *(const float4*)(H + (size_t)s * C + f);
            float* ap = agg + (size_t)d * C + f;
            atomicAdd(ap + 0, v.x * alpha);
            atomicAdd(ap + 1, v.y * alpha);
            atomicAdd(ap + 2, v.z * alpha);
            atomicAdd(ap + 3, v.w * alpha);
        }
    }
}

// ---------------------------------------------------------------------------
// BatchNorm (training-style, stats over rows) + ReLU, optional bf16 mirror.
// Note: additive bias before BN is an exact no-op through mean subtraction.
// ---------------------------------------------------------------------------
__global__ void bn_stats_k(const float* __restrict__ X, float* __restrict__ stats,
                           int n, int C, int rowsPer) {
    int c = blockIdx.x * blockDim.x + threadIdx.x;   // blockDim = 128
    if (c >= C) return;
    int r0 = blockIdx.y * rowsPer;
    int r1 = min(n, r0 + rowsPer);
    float s = 0.f, s2 = 0.f;
    for (int r = r0; r < r1; ++r) {
        float v = X[(size_t)r * C + c];
        s += v; s2 += v * v;
    }
    atomicAdd(&stats[c], s);
    atomicAdd(&stats[C + c], s2);
}

__global__ void bn_act_k(float* __restrict__ X, const float* __restrict__ stats,
                         const float* __restrict__ g, const float* __restrict__ be,
                         unsigned short* __restrict__ xbf, int n, int C) {
    long i = (long)blockIdx.x * blockDim.x + threadIdx.x;
    long tot = (long)n * C;
    if (i >= tot) return;
    int c = (int)(i % C);
    float inv_n = 1.0f / (float)n;
    float mu  = stats[c] * inv_n;
    float var = stats[C + c] * inv_n - mu * mu;
    float y = (X[i] - mu) * rsqrtf(var + 1e-5f) * g[c] + be[c];
    y = y > 0.f ? y : 0.f;
    X[i] = y;
    if (xbf) xbf[i] = f2bf(y);
}

// ---------------------------------------------------------------------------
// Pooling and MLP head (tiny: 64 rows)
// ---------------------------------------------------------------------------
__global__ void pool_sum_k(const float* __restrict__ X, const int* __restrict__ batch,
                           float* __restrict__ pool, float* __restrict__ cnt, int n) {
    int nid = blockIdx.x;           // grid = N, block = 128
    int c = threadIdx.x;
    if (nid >= n) return;
    int g = batch[nid];
    atomicAdd(&pool[(size_t)g * 128 + c], X[(size_t)nid * 128 + c]);
    if (c == 0) atomicAdd(&cnt[g], 1.0f);
}

__global__ void concat_pool_k(const float* __restrict__ pool, const float* __restrict__ cnt,
                              const float* __restrict__ gf, float* __restrict__ z) {
    int i = blockIdx.x * blockDim.x + threadIdx.x;   // 64*192
    if (i >= 64 * 192) return;
    int g = i / 192, c = i % 192;
    z[i] = (c < 128) ? pool[g * 128 + c] / fmaxf(cnt[g], 1.0f)
                     : gf[g * 64 + (c - 128)];
}

__global__ void small_gemm_bias_k(const float* __restrict__ A, const float* __restrict__ B,
                                  const float* __restrict__ bias, float* __restrict__ C,
                                  int M, int K, int N) {
    int i = blockIdx.x * blockDim.x + threadIdx.x;
    if (i >= M * N) return;
    int mm = i / N, nn = i % N;
    float s = bias ? bias[nn] : 0.f;
    for (int k = 0; k < K; ++k) s += A[mm * K + k] * B[k * N + nn];
    C[i] = s;
}

// ---------------------------------------------------------------------------
// Host launch
// ---------------------------------------------------------------------------
extern "C" void kernel_launch(void* const* d_in, const int* in_sizes, int n_in,
                              void* d_out, int out_size, void* d_ws, size_t ws_size,
                              hipStream_t stream) {
    const float* x     = (const float*)d_in[0];
    const int*   ei    = (const int*)d_in[1];
    const int*   batch = (const int*)d_in[2];
    const float* gfeat = (const float*)d_in[3];
    const float* W1  = (const float*)d_in[4];
    const float* a1s = (const float*)d_in[5];
    const float* a1d = (const float*)d_in[6];
    const float* g1  = (const float*)d_in[8];
    const float* be1 = (const float*)d_in[9];
    const float* W2  = (const float*)d_in[10];
    const float* a2s = (const float*)d_in[11];
    const float* a2d = (const float*)d_in[12];
    const float* g2  = (const float*)d_in[14];
    const float* be2 = (const float*)d_in[15];
    const float* W3  = (const float*)d_in[16];
    const float* a3s = (const float*)d_in[17];
    const float* a3d = (const float*)d_in[18];
    const float* g3  = (const float*)d_in[20];
    const float* be3 = (const float*)d_in[21];
    const float* fw1 = (const float*)d_in[22];
    const float* fb1 = (const float*)d_in[23];
    const float* gf1 = (const float*)d_in[24];
    const float* bef1= (const float*)d_in[25];
    const float* fw2 = (const float*)d_in[26];
    const float* fb2 = (const float*)d_in[27];
    const float* gf2 = (const float*)d_in[28];
    const float* bef2= (const float*)d_in[29];
    const float* fw3 = (const float*)d_in[30];
    const float* fb3 = (const float*)d_in[31];

    const int N = in_sizes[0] / 128;         // 20000
    const int E = in_sizes[1] / 2;           // 160000
    const int Etot = E + N;                  // + self loops

    // workspace layout
    char* base = (char*)d_ws;
    size_t off = 0;
    auto take = [&](size_t bytes) -> void* {
        void* p = base + off;
        off += (bytes + 255) & ~(size_t)255;
        return p;
    };
    unsigned short* XBF = (unsigned short*)take((size_t)N * 512 * 2);
    unsigned short* WBF = (unsigned short*)take((size_t)512 * 512 * 2);
    float* Hbuf = (float*)take((size_t)N * 512 * 4);
    float* AGG  = (float*)take((size_t)N * 512 * 4);
    float* ASRC = (float*)take((size_t)N * 4 * 4);
    float* ADST = (float*)take((size_t)N * 4 * 4);
    float* MM   = (float*)take((size_t)N * 4 * 4);
    float* DEN  = (float*)take((size_t)N * 4 * 4);
    float* STATS= (float*)take(1024 * 4);
    float* POOL = (float*)take(64 * 128 * 4);
    float* CNT  = (float*)take(64 * 4);
    float* Z0   = (float*)take(64 * 192 * 4);
    float* Z1   = (float*)take(64 * 256 * 4);
    float* Z2   = (float*)take(64 * 128 * 4);

    auto cdiv = [](long a, long b) { return (int)((a + b - 1) / b); };

    auto gat_layer = [&](int K, int heads, const float* W,
                         const float* as, const float* ad,
                         const float* g, const float* be,
                         unsigned short* outBf) {
        const int C = heads * 128;
        // weights -> bf16, transposed to [Nout][K] for b128-chunk staging
        cast_bf16_t_k<<<cdiv((long)K * C, 256), 256, 0, stream>>>(W, WBF, K, C);
        // H = XBF @ WBF^T  (WMMA)
        dim3 gg(cdiv(N, 64), C / 64);
        gemm_bf16_wmma<<<gg, 128, 0, stream>>>(XBF, WBF, Hbuf, N, K, C);
        // attention logits per node
        attn_scores_k<<<cdiv((long)N * 32, 256), 256, 0, stream>>>(Hbuf, as, ad, ASRC, ADST, N, heads);
        // segment softmax over dst
        init_softmax_k<<<cdiv((long)N * heads, 256), 256, 0, stream>>>(MM, DEN, N * heads);
        edge_max_k<<<cdiv(Etot, 256), 256, 0, stream>>>(ei, E, Etot, ASRC, ADST, MM, heads);
        edge_sum_k<<<cdiv(Etot, 256), 256, 0, stream>>>(ei, E, Etot, ASRC, ADST, MM, DEN, heads);
        // aggregate messages
        zero_f32_k<<<2048, 256, 0, stream>>>(AGG, (long)N * C);
        edge_agg_k<<<1024, 256, 0, stream>>>(ei, E, Etot, Hbuf, ASRC, ADST, MM, DEN, AGG, heads, C);
        // BN + ReLU (bias cancels through BN mean subtraction)
        zero_f32_k<<<cdiv(2L * C, 256), 256, 0, stream>>>(STATS, 2L * C);
        int rows = 500;
        dim3 sb(C / 128, cdiv(N, rows));
        bn_stats_k<<<sb, 128, 0, stream>>>(AGG, STATS, N, C, rows);
        bn_act_k<<<cdiv((long)N * C, 256), 256, 0, stream>>>(AGG, STATS, g, be, outBf, N, C);
    };

    // Layer 1: in=128 -> 4x128 concat
    cast_bf16_k<<<cdiv((long)N * 128, 256), 256, 0, stream>>>(x, XBF, (long)N * 128);
    gat_layer(128, 4, W1, a1s, a1d, g1, be1, XBF);
    // Layer 2: in=512 -> 4x128 concat
    gat_layer(512, 4, W2, a2s, a2d, g2, be2, XBF);
    // Layer 3: in=512 -> 1x128 (mean over 1 head == identity)
    gat_layer(512, 1, W3, a3s, a3d, g3, be3, nullptr);

    // global mean pool + concat global features
    zero_f32_k<<<cdiv(64L * 128, 256), 256, 0, stream>>>(POOL, 64L * 128);
    zero_f32_k<<<1, 64, 0, stream>>>(CNT, 64);
    pool_sum_k<<<N, 128, 0, stream>>>(AGG, batch, POOL, CNT, N);
    concat_pool_k<<<cdiv(64L * 192, 256), 256, 0, stream>>>(POOL, CNT, gfeat, Z0);

    // MLP head (64 rows; naive fp32 is plenty)
    small_gemm_bias_k<<<cdiv(64L * 256, 256), 256, 0, stream>>>(Z0, fw1, fb1, Z1, 64, 192, 256);
    zero_f32_k<<<2, 256, 0, stream>>>(STATS, 512);
    {
        dim3 sb(256 / 128, 1);
        bn_stats_k<<<sb, 128, 0, stream>>>(Z1, STATS, 64, 256, 64);
        bn_act_k<<<cdiv(64L * 256, 256), 256, 0, stream>>>(Z1, STATS, gf1, bef1, nullptr, 64, 256);
    }
    small_gemm_bias_k<<<cdiv(64L * 128, 256), 256, 0, stream>>>(Z1, fw2, fb2, Z2, 64, 256, 128);
    zero_f32_k<<<1, 256, 0, stream>>>(STATS, 256);
    {
        dim3 sb(128 / 128, 1);
        bn_stats_k<<<sb, 128, 0, stream>>>(Z2, STATS, 64, 128, 64);
        bn_act_k<<<cdiv(64L * 128, 256), 256, 0, stream>>>(Z2, STATS, gf2, bef2, nullptr, 64, 128);
    }
    small_gemm_bias_k<<<cdiv(64L * 16, 256), 256, 0, stream>>>(Z2, fw3, fb3, (float*)d_out, 64, 128, 16);

    (void)n_in; (void)out_size; (void)ws_size;
}
